// TransportModule_24592982737605
// MI455X (gfx1250) — matrix operand order repaired and checked by reading, not compiled, not run on hardware
//
#include <hip/hip_runtime.h>
#include <hip/hip_bf16.h>

typedef __attribute__((ext_vector_type(16))) _Float16 v16h;
typedef __attribute__((ext_vector_type(8)))  _Float16 v8h;
typedef __attribute__((ext_vector_type(8)))  float    v8f;

#define NPTS    8192
#define PCTRS   256
#define WAVES_A 8

// 16-bit B fragment (32x16, KxN): k held in half h of this lane
__device__ __forceinline__ int b_k_of(int lane, int h) {
  return (h < 8) ? ((lane >> 4) << 3) + h : 16 + ((lane >> 4) << 3) + (h - 8);
}
__device__ __forceinline__ v8f wmma_f16(v16h a, v16h b, v8f c) {
  return __builtin_amdgcn_wmma_f32_16x16x32_f16(false, a, false, b, (short)0, c,
                                                false, false);
}
// Branch-free tanh: gfx1250 hardware v_tanh_f32 (confirmed lowering).
__device__ __forceinline__ float fast_tanh(float x) {
#if __has_builtin(__builtin_amdgcn_tanhf)
  return __builtin_amdgcn_tanhf(x);
#else
  float e = __builtin_amdgcn_exp2f(x * 2.885390081777927f);   // 2*log2(e)
  return 1.f - 2.f * __builtin_amdgcn_rcpf(e + 1.f);
#endif
}
__device__ __forceinline__ float fast_exp(float x) {
  return __builtin_amdgcn_exp2f(x * 1.4426950408889634f);     // log2(e)
}
__device__ __forceinline__ float fast_rcp(float x)  { return __builtin_amdgcn_rcpf(x); }
__device__ __forceinline__ float fast_sqrt(float x) { return __builtin_amdgcn_sqrtf(x); }
__device__ __forceinline__ float fast_rsq(float x)  { return __builtin_amdgcn_rsqf(x); }

// A-operand (16x32 MxK) from a row-major LDS activation tile.
// rowp must already include mrow*stride + grp*8; koff is the K-tile base.
// halves 0..7 <-> k = koff+grp*8+(0..7); halves 8..15 <-> +16.
__device__ __forceinline__ v16h load_a_frag(const _Float16* rowp, int koff) {
  v8h lo = *(const v8h*)(rowp + koff);
  v8h hi = *(const v8h*)(rowp + koff + 16);
  return __builtin_shufflevector(lo, hi, 0, 1, 2, 3, 4, 5, 6, 7,
                                 8, 9, 10, 11, 12, 13, 14, 15);
}

// ---------------------------------------------------------------------------
// Kernel A: CoefficientNet. One block (8 wave32) per particle n.
// ---------------------------------------------------------------------------
__global__ __launch_bounds__(256)
void attn_coeff_kernel(const float* __restrict__ x, const float* __restrict__ v,
                       const float* __restrict__ cpos, const float* __restrict__ cval,
                       const float* __restrict__ W1, const float* __restrict__ B1,
                       const float* __restrict__ W2, const float* __restrict__ B2,
                       const float* __restrict__ W3, const float* __restrict__ B3,
                       float* __restrict__ coeff_out)
{
  __shared__ v16h     sW1f[4 * 32];              //  4 KB  (6->64, K pad 32)
  __shared__ v16h     sW2f[8 * 32];              //  8 KB  (64->64, 2 K-tiles x 4 N-tiles)
  __shared__ _Float16 out1[WAVES_A * 16 * 64];   // 16 KB  per-wave activation tile
  __shared__ float    wred[WAVES_A][16][16];     // 8 KB  head reduction
  __shared__ float    slog[PCTRS];
  __shared__ float    spos[PCTRS];
  __shared__ float    sred[256];
  __shared__ float    sW3[64], sB1v[64], sB2v[64];

  const int tid = threadIdx.x;

  // Stage weights in fragment-swizzled order (B-operand ready).
  {
    _Float16* d = (_Float16*)sW1f;
    for (int i = tid; i < 4 * 32 * 16; i += 256) {
      int h = i & 15, lane = (i >> 4) & 31, nt = i >> 9;
      int k = b_k_of(lane, h), n = nt * 16 + (lane & 15);
      d[i] = (_Float16)((k < 6) ? W1[k * 64 + n] : 0.f);
    }
  }
  {
    _Float16* d = (_Float16*)sW2f;
    for (int i = tid; i < 8 * 32 * 16; i += 256) {
      int h = i & 15, lane = (i >> 4) & 31, f = i >> 9;
      int kt = f >> 2, nt = f & 3;
      int k = kt * 32 + b_k_of(lane, h), n = nt * 16 + (lane & 15);
      d[i] = (_Float16)W2[k * 64 + n];
    }
  }
  if (tid < 64) { sW3[tid] = W3[tid]; sB1v[tid] = B1[tid]; sB2v[tid] = B2[tid]; }
  __syncthreads();

  const int   n_idx = blockIdx.x;
  const float x0 = x[2 * n_idx], x1 = x[2 * n_idx + 1];
  const float v0 = v[2 * n_idx], v1 = v[2 * n_idx + 1];
  const float inv_vn = fast_rsq(v0 * v0 + v1 * v1 + 1e-16f);
  const float ag0 = v0 * inv_vn, ag1 = v1 * inv_vn;
  const float b3v = B3[0];

  const int wave = tid >> 5, lane = tid & 31;
  const int mrow = lane & 15, grp = lane >> 4;
  _Float16* ob = out1 + wave * (16 * 64);
  const _Float16* rowp = ob + mrow * 64 + grp * 8;   // A-frag base for this lane

  for (int t = wave; t < 16; t += WAVES_A) {     // 16 row-tiles of 16 pairs
    const int p  = t * 16 + mrow;
    float cx = cpos[2 * p], cy = cpos[2 * p + 1];
    float rx = x0 - cx, ry = x1 - cy;
    float dist = fast_sqrt(rx * rx + ry * ry + 1e-16f);
    float pos  = rx * ag0 + ry * ag1;
    float ang  = pos * fast_rcp(dist + 1e-8f);
    if (lane < 16) spos[p] = pos;
    float feat[6] = {x0, x1, v0, v1, ang, pos};

    // A1: k = grp*8 + h for h<8 (k>=16 for h>=8 -> zero pad).
    // Only grp==0 lanes carry live k (0..5).
    v16h a1 = {};
    #pragma unroll
    for (int h = 0; h < 6; ++h)
      a1[h] = (grp == 0) ? (_Float16)feat[h] : (_Float16)0.f;

    // Layer 1: 16x64 = 4 WMMA; bias + tanh -> LDS activation tile.
    #pragma unroll
    for (int nt = 0; nt < 4; ++nt) {
      v8f acc = {};
      acc = wmma_f16(a1, sW1f[nt * 32 + lane], acc);
      int nn = nt * 16 + mrow;
      float bb = sB1v[nn];
      #pragma unroll
      for (int r = 0; r < 8; ++r) {
        int mm = r + 8 * grp;
        ob[mm * 64 + nn] = (_Float16)fast_tanh(acc[r] + bb);
      }
    }
    asm volatile("s_wait_dscnt 0" ::: "memory");
    // Layer 2 A-operands via two ds_load_b128 each.
    v16h a2lo = load_a_frag(rowp, 0);
    v16h a2hi = load_a_frag(rowp, 32);

    float part[8] = {0, 0, 0, 0, 0, 0, 0, 0};
    #pragma unroll
    for (int nt = 0; nt < 4; ++nt) {
      v8f acc = {};
      acc = wmma_f16(a2lo, sW2f[(0 * 4 + nt) * 32 + lane], acc);
      acc = wmma_f16(a2hi, sW2f[(1 * 4 + nt) * 32 + lane], acc);
      int nn = nt * 16 + mrow;
      float w3 = sW3[nn], bb = sB2v[nn];
      #pragma unroll
      for (int r = 0; r < 8; ++r)
        part[r] += fast_tanh(acc[r] + bb) * w3;
    }
    // Head: reduce 64-wide dot across lanes through LDS.
    #pragma unroll
    for (int r = 0; r < 8; ++r) wred[wave][grp * 8 + r][mrow] = part[r];
    asm volatile("s_wait_dscnt 0" ::: "memory");
    if (lane < 16) {
      float s = 0.f;
      #pragma unroll
      for (int j = 0; j < 16; ++j) s += wred[wave][lane][j];
      int pp = t * 16 + lane;
      float logit = s + b3v;
      slog[pp] = (spos[pp] > 0.f) ? logit : -1e30f;
    }
  }
  __syncthreads();

  // Masked softmax over P=256, then attn @ coeff_values, coeff = exp(-attn).
  float l = slog[tid];
  sred[tid] = l; __syncthreads();
  for (int s = 128; s > 0; s >>= 1) {
    if (tid < s) sred[tid] = fmaxf(sred[tid], sred[tid + s]);
    __syncthreads();
  }
  float mx = sred[0]; __syncthreads();
  float e = fast_exp(l - mx);
  sred[tid] = e; __syncthreads();
  for (int s = 128; s > 0; s >>= 1) {
    if (tid < s) sred[tid] += sred[tid + s];
    __syncthreads();
  }
  float w = e * fast_rcp(sred[0]); __syncthreads();

  float a0 = w * cval[2 * tid], a1c = w * cval[2 * tid + 1];
  sred[tid] = a0; __syncthreads();
  for (int s = 128; s > 0; s >>= 1) {
    if (tid < s) sred[tid] += sred[tid + s];
    __syncthreads();
  }
  if (tid == 0) coeff_out[2 * n_idx] = fast_exp(-sred[0]);
  __syncthreads();
  sred[tid] = a1c; __syncthreads();
  for (int s = 128; s > 0; s >>= 1) {
    if (tid < s) sred[tid] += sred[tid + s];
    __syncthreads();
  }
  if (tid == 0) coeff_out[2 * n_idx + 1] = fast_exp(-sred[0]);
}

// ---------------------------------------------------------------------------
// Kernel B: TransportModule. 64 blocks x 8 waves x 16 rows.
// Full 256x256 W2 staged fragment-swizzled in LDS (128 KB of 320 KB/WGP).
// ---------------------------------------------------------------------------
constexpr int W1F_BYTES  = 16 * 32 * 32;          //  16 KB
constexpr int W2F_BYTES  = 128 * 32 * 32;         // 128 KB
constexpr int OUT1_BYTES = 128 * 256 * 2;         //  64 KB
constexpr int WRED_BYTES = 8 * 16 * 16 * 4;       //  8 KB
constexpr int SMEM_T = W1F_BYTES + W2F_BYTES + OUT1_BYTES + WRED_BYTES
                     + 3 * 256 * (int)sizeof(float);

__global__ __launch_bounds__(256)
void transport_kernel(const float* __restrict__ x,  const float* __restrict__ v,
                      const float* __restrict__ xp, const float* __restrict__ vp,
                      const float* __restrict__ coeff,
                      const float* __restrict__ W1, const float* __restrict__ B1,
                      const float* __restrict__ W2, const float* __restrict__ B2,
                      const float* __restrict__ W3, const float* __restrict__ B3,
                      float* __restrict__ out)
{
  extern __shared__ char smem[];
  v16h*     w1f  = (v16h*)smem;
  v16h*     w2f  = (v16h*)(smem + W1F_BYTES);
  _Float16* out1 = (_Float16*)(smem + W1F_BYTES + W2F_BYTES);
  float*    wred = (float*)(smem + W1F_BYTES + W2F_BYTES + OUT1_BYTES);
  float*    sb1  = (float*)(smem + W1F_BYTES + W2F_BYTES + OUT1_BYTES + WRED_BYTES);
  float*    sb2  = sb1 + 256;
  float*    sw3  = sb2 + 256;

  const int tid = threadIdx.x;
  {
    _Float16* d = (_Float16*)w1f;
    for (int i = tid; i < 16 * 32 * 16; i += 256) {
      int h = i & 15, lane = (i >> 4) & 31, nt = i >> 9;
      int k = b_k_of(lane, h), n = nt * 16 + (lane & 15);
      d[i] = (_Float16)((k < 10) ? W1[k * 256 + n] : 0.f);
    }
  }
  {
    _Float16* d = (_Float16*)w2f;
    for (int i = tid; i < 128 * 32 * 16; i += 256) {
      int h = i & 15, lane = (i >> 4) & 31, f = i >> 9;
      int kt = f >> 4, nt = f & 15;
      int k = kt * 32 + b_k_of(lane, h), n = nt * 16 + (lane & 15);
      d[i] = (_Float16)W2[k * 256 + n];
    }
  }
  sb1[tid] = B1[tid]; sb2[tid] = B2[tid]; sw3[tid] = W3[tid];
  __syncthreads();

  const int wave = tid >> 5, lane = tid & 31;
  const int mrow = lane & 15, grp = lane >> 4;
  const int rowbase = blockIdx.x * 128 + wave * 16;
  const float b3v = B3[0];
  _Float16* ob = out1 + wave * 16 * 256;
  const _Float16* rowp = ob + mrow * 256 + grp * 8;  // A-frag base for this lane

  const int row = rowbase + mrow;
  float feat[10] = { x[2 * row],  x[2 * row + 1],  v[2 * row],  v[2 * row + 1],
                     xp[2 * row], xp[2 * row + 1], vp[2 * row], vp[2 * row + 1],
                     coeff[2 * row], coeff[2 * row + 1] };
  // A1: h<8 -> k = grp*8 + h. grp0 carries k 0..7, grp1 carries k 8..9 (h<2).
  // h>=8 -> k>=16 -> zero pad.
  v16h a1 = {};
  #pragma unroll
  for (int h = 0; h < 8; ++h) {
    float lo = feat[h];
    float hi = (h < 2) ? feat[8 + h] : 0.f;
    a1[h] = (_Float16)(grp ? hi : lo);
  }
  // Layer 1: 16 WMMA (N=256), bias + tanh -> LDS.
  for (int nt = 0; nt < 16; ++nt) {
    v8f acc = {};
    acc = wmma_f16(a1, w1f[nt * 32 + lane], acc);
    int nn = nt * 16 + mrow;
    float bb = sb1[nn];
    #pragma unroll
    for (int r = 0; r < 8; ++r) {
      int mm = r + 8 * grp;
      ob[mm * 256 + nn] = (_Float16)fast_tanh(acc[r] + bb);
    }
  }
  asm volatile("s_wait_dscnt 0" ::: "memory");
  // Layer 2 A-operands: 8 K-tiles resident in registers (2 ds_load_b128 each).
  v16h a2[8];
  #pragma unroll
  for (int kt = 0; kt < 8; ++kt)
    a2[kt] = load_a_frag(rowp, kt * 32);

  float part[8] = {0, 0, 0, 0, 0, 0, 0, 0};
  for (int nt = 0; nt < 16; ++nt) {            // 128 WMMA: 8-deep K chains
    v8f acc = {};
    #pragma unroll
    for (int kt = 0; kt < 8; ++kt)
      acc = wmma_f16(a2[kt], w2f[(kt * 16 + nt) * 32 + lane], acc);
    int nn = nt * 16 + mrow;
    float w3 = sw3[nn], bb = sb2[nn];
    #pragma unroll
    for (int r = 0; r < 8; ++r)
      part[r] += fast_tanh(acc[r] + bb) * w3;
  }
  #pragma unroll
  for (int r = 0; r < 8; ++r)
    wred[(wave * 16 + grp * 8 + r) * 16 + mrow] = part[r];
  asm volatile("s_wait_dscnt 0" ::: "memory");
  if (lane < 16) {
    float s = 0.f;
    #pragma unroll
    for (int j = 0; j < 16; ++j) s += wred[(wave * 16 + lane) * 16 + j];
    out[rowbase + lane] = fast_exp(fast_tanh(s + b3v));
  }
}

// ---------------------------------------------------------------------------
extern "C" void kernel_launch(void* const* d_in, const int* in_sizes, int n_in,
                              void* d_out, int out_size, void* d_ws, size_t ws_size,
                              hipStream_t stream) {
  const float* x   = (const float*)d_in[0];
  const float* v   = (const float*)d_in[1];
  const float* xp  = (const float*)d_in[2];
  const float* vp  = (const float*)d_in[3];
  const float* cp  = (const float*)d_in[4];
  const float* cv  = (const float*)d_in[5];
  const float* aW1 = (const float*)d_in[6];
  const float* ab1 = (const float*)d_in[7];
  const float* aW2 = (const float*)d_in[8];
  const float* ab2 = (const float*)d_in[9];
  const float* aW3 = (const float*)d_in[10];
  const float* ab3 = (const float*)d_in[11];
  const float* tW1 = (const float*)d_in[12];
  const float* tb1 = (const float*)d_in[13];
  const float* tW2 = (const float*)d_in[14];
  const float* tb2 = (const float*)d_in[15];
  const float* tW3 = (const float*)d_in[16];
  const float* tb3 = (const float*)d_in[17];
  float* coeff = (float*)d_ws;                 // N x C scratch
  float* out   = (float*)d_out;

  attn_coeff_kernel<<<NPTS, 256, 0, stream>>>(x, v, cp, cv, aW1, ab1, aW2, ab2,
                                              aW3, ab3, coeff);

  (void)hipFuncSetAttribute((const void*)transport_kernel,
                            hipFuncAttributeMaxDynamicSharedMemorySize, SMEM_T);
  transport_kernel<<<NPTS / 128, 256, SMEM_T, stream>>>(x, v, xp, vp, coeff,
                                                        tW1, tb1, tW2, tb2,
                                                        tW3, tb3, out);
}
